// MyNet_12094627905538
// MI455X (gfx1250) — matrix operand order repaired and checked by reading, not compiled, not run on hardware
//
#include <hip/hip_runtime.h>
#include <hip/hip_bf16.h>
#include <math.h>

typedef __attribute__((ext_vector_type(16))) _Float16 v16h;
typedef __attribute__((ext_vector_type(8)))  _Float16 v8h;
typedef __attribute__((ext_vector_type(8)))  float    v8f;

// ---------------------------------------------------------------------------
// Problem constants
//   input x: (8,3,256,256) f32, y: (8,256,256,8) f32 NHWC
//   Ho=Wo=128, samples M = 8*128*128 = 131072, Kdim = 72 (padded to 96)
//   out: (8,512,128,128) f32
// ---------------------------------------------------------------------------

__device__ __forceinline__ v16h ld2x8(const _Float16* p0, const _Float16* p1) {
  union { v16h v; v8h h[2]; } u;
  u.h[0] = *(const v8h*)p0;
  u.h[1] = *(const v8h*)p1;
  return u.v;
}

// Kernel 1: 1x1 conv (3->8) + bias + LeakyReLU, NCHW in -> NHWC out
__global__ void __launch_bounds__(256) conv1_lrelu(const float* __restrict__ x,
                                                   const float* __restrict__ w1,
                                                   const float* __restrict__ b1,
                                                   float* __restrict__ y) {
  const int idx = blockIdx.x * 256 + threadIdx.x;     // 524288 = 8*256*256
  const int hw  = idx & 65535;
  const int n   = idx >> 16;
  const float* xp = x + (size_t)n * 3 * 65536 + hw;
  const float x0 = xp[0], x1 = xp[65536], x2 = xp[131072];
  float ov[8];
#pragma unroll
  for (int o = 0; o < 8; ++o) {
    float v = b1[o] + w1[3*o]*x0 + w1[3*o+1]*x1 + w1[3*o+2]*x2;
    ov[o] = (v >= 0.f) ? v : 0.1f * v;
  }
  float4* yp = (float4*)(y + (size_t)idx * 8);
  yp[0] = make_float4(ov[0], ov[1], ov[2], ov[3]);
  yp[1] = make_float4(ov[4], ov[5], ov[6], ov[7]);
}

// Kernel 2: cast dcn_w (512x72 f32, flat o*72 + c*9 + k) to f16 padded 512x96
__global__ void __launch_bounds__(256) wcast(const float* __restrict__ dw,
                                             _Float16* __restrict__ wf) {
  const int i = blockIdx.x * 256 + threadIdx.x;       // 49152 = 512*96
  const int o = i / 96, k = i % 96;
  wf[i] = (k < 72) ? (_Float16)dw[o*72 + k] : (_Float16)0.f;
}

// Kernel 3: offset conv (3x3 s2 p1, 8->27) + deformable bilinear sampling.
// One thread per output position; emits one 96-wide f16 cols row
// (layout c*9+k, cols 72..95 zero) via packed v8h stores.
__global__ void __launch_bounds__(256) dcn_sample(const float* __restrict__ y,
                                                  const float* __restrict__ offw,
                                                  const float* __restrict__ offb,
                                                  _Float16* __restrict__ cols) {
  __shared__ float sw[27 * 72];
  __shared__ float sbias[27];
  for (int i = threadIdx.x; i < 27 * 72; i += 256) sw[i] = offw[i];
  if (threadIdx.x < 27) sbias[threadIdx.x] = offb[threadIdx.x];
  __syncthreads();

  const int idx = blockIdx.x * 256 + threadIdx.x;     // 131072 = 8*128*128
  const int w = idx & 127;
  const int h = (idx >> 7) & 127;
  const int n = idx >> 14;

  float om[27];
#pragma unroll
  for (int oc = 0; oc < 27; ++oc) om[oc] = sbias[oc];

  // offset conv: om[oc] += sum_{cin,t} y[n,hh,ww,cin] * offw[oc,cin,t]
  for (int t = 0; t < 9; ++t) {
    const int ky = t / 3, kx = t % 3;
    const int hh = 2*h - 1 + ky;
    const int ww = 2*w - 1 + kx;
    if (hh < 0 || hh > 255 || ww < 0 || ww > 255) continue;
    const float4* yp = (const float4*)(y + (((size_t)n * 256 + hh) * 256 + ww) * 8);
    const float4 va = yp[0], vb = yp[1];
    const float yv[8] = {va.x, va.y, va.z, va.w, vb.x, vb.y, vb.z, vb.w};
#pragma unroll
    for (int oc = 0; oc < 27; ++oc) {
      float s = om[oc];
#pragma unroll
      for (int c = 0; c < 8; ++c) s += yv[c] * sw[oc*72 + c*9 + t];
      om[oc] = s;
    }
  }

  _Float16 row[96];
#pragma unroll
  for (int j = 72; j < 96; ++j) row[j] = (_Float16)0.f;

#pragma unroll
  for (int k = 0; k < 9; ++k) {
    const int ky = k / 3, kx = k % 3;
    const float py = (float)(2*h - 1 + ky) + om[2*k];
    const float px = (float)(2*w - 1 + kx) + om[2*k + 1];
    const float mk = 1.f / (1.f + __expf(-om[18 + k]));
    const float y0 = floorf(py);
    const float x0 = floorf(px);
    float acc[8] = {0.f, 0.f, 0.f, 0.f, 0.f, 0.f, 0.f, 0.f};
#pragma unroll
    for (int oy = 0; oy < 2; ++oy) {
#pragma unroll
      for (int ox = 0; ox < 2; ++ox) {
        const float yy = y0 + (float)oy;
        const float xx = x0 + (float)ox;
        const float wt = (1.f - fabsf(py - yy)) * (1.f - fabsf(px - xx));
        if (yy >= 0.f && yy < 256.f && xx >= 0.f && xx < 256.f) {
          const int iy = (int)yy, ix = (int)xx;
          const float4* sp = (const float4*)(y + (((size_t)n * 256 + iy) * 256 + ix) * 8);
          const float4 sa = sp[0], sb4 = sp[1];
          acc[0] += wt * sa.x;  acc[1] += wt * sa.y;
          acc[2] += wt * sa.z;  acc[3] += wt * sa.w;
          acc[4] += wt * sb4.x; acc[5] += wt * sb4.y;
          acc[6] += wt * sb4.z; acc[7] += wt * sb4.w;
        }
      }
    }
#pragma unroll
    for (int c = 0; c < 8; ++c) row[c*9 + k] = (_Float16)(acc[c] * mk);
  }

  v8h* cp = (v8h*)(cols + (size_t)idx * 96);
#pragma unroll
  for (int j = 0; j < 12; ++j) {
    v8h tv;
#pragma unroll
    for (int e = 0; e < 8; ++e) tv[e] = row[j*8 + e];
    cp[j] = tv;
  }
}

// Kernel 4: WMMA GEMM  D(512 x 131072) = W(512x96) x cols^T(96x131072)
// One wave: 16 spatial cols x 64 output channels (4 accumulators),
// 3 k-steps of v_wmma_f32_16x16x32_f16 per accumulator = 12 WMMA/wave.
// A (16-bit 16x32): lane holds row o; two 8-half runs at K {kA..kA+7, kA+16..kA+23},
//   kA = 0/8 by lane half.  B (32x16): lane holds column m; two 8-half runs at
//   K {kB..kB+7, kB+16..kB+23}, kB = 0/8 by lane half.  D: N(=m) across lanes, M(=o) across VGPRs (+8 hi half)
// -> stores are two 64B-contiguous runs per instruction into NCHW output.
__global__ void __launch_bounds__(256) dcn_gemm(const _Float16* __restrict__ cols,
                                                const _Float16* __restrict__ wgt,
                                                float* __restrict__ out) {
  const int lane  = threadIdx.x & 31;
  const int wv    = threadIdx.x >> 5;          // 0..7
  const int ln    = lane & 15;
  const bool hiH  = (lane >= 16);
  const int m     = blockIdx.x * 16 + ln;      // spatial sample index
  const int obase = wv * 64;

  const int kA = hiH ? 8 : 0;
  const int kB = hiH ? 8 : 0;

  const _Float16* bp  = cols + (size_t)m * 96 + kB;
  const _Float16* ap0 = wgt + (size_t)(obase + ln) * 96 + kA;
  const _Float16* ap1 = ap0 + 16 * 96;
  const _Float16* ap2 = ap0 + 32 * 96;
  const _Float16* ap3 = ap0 + 48 * 96;

  v8f c0 = {}, c1 = {}, c2 = {}, c3 = {};
#pragma unroll
  for (int kb = 0; kb < 96; kb += 32) {
    const v16h b  = ld2x8(bp  + kb, bp  + kb + 16);
    const v16h a0 = ld2x8(ap0 + kb, ap0 + kb + 16);
    const v16h a1 = ld2x8(ap1 + kb, ap1 + kb + 16);
    const v16h a2 = ld2x8(ap2 + kb, ap2 + kb + 16);
    const v16h a3 = ld2x8(ap3 + kb, ap3 + kb + 16);
    c0 = __builtin_amdgcn_wmma_f32_16x16x32_f16(false, a0, false, b, (short)0, c0, false, false);
    c1 = __builtin_amdgcn_wmma_f32_16x16x32_f16(false, a1, false, b, (short)0, c1, false, false);
    c2 = __builtin_amdgcn_wmma_f32_16x16x32_f16(false, a2, false, b, (short)0, c2, false, false);
    c3 = __builtin_amdgcn_wmma_f32_16x16x32_f16(false, a3, false, b, (short)0, c3, false, false);
  }

  const int n  = m >> 14;                       // 16384 spatial per image
  const int hw = m & 16383;
  float* o0 = out + (((size_t)(n * 512 + obase) + (hiH ? 8 : 0)) << 14) + hw;
#pragma unroll
  for (int r = 0; r < 8; ++r) {
    o0[(size_t)(r     ) << 14] = c0[r];
    o0[(size_t)(r + 16) << 14] = c1[r];
    o0[(size_t)(r + 32) << 14] = c2[r];
    o0[(size_t)(r + 48) << 14] = c3[r];
  }
}

extern "C" void kernel_launch(void* const* d_in, const int* in_sizes, int n_in,
                              void* d_out, int out_size, void* d_ws, size_t ws_size,
                              hipStream_t stream) {
  const float* x  = (const float*)d_in[0];   // (8,3,256,256)
  const float* w1 = (const float*)d_in[1];   // (8,3,1,1)
  const float* b1 = (const float*)d_in[2];   // (8,)
  const float* ow = (const float*)d_in[3];   // (27,8,3,3)
  const float* ob = (const float*)d_in[4];   // (27,)
  const float* dw = (const float*)d_in[5];   // (512,8,3,3)
  float* out = (float*)d_out;                // (8,512,128,128)

  char* ws = (char*)d_ws;
  float*    y    = (float*)ws;                                   // 16 MiB  NHWC activations
  _Float16* cols = (_Float16*)(ws + (size_t)16777216);           // 24 MiB  131072 x 96 f16
  _Float16* wf   = (_Float16*)(ws + (size_t)16777216 + 25165824);// 96 KiB  512 x 96 f16

  conv1_lrelu<<<2048, 256, 0, stream>>>(x, w1, b1, y);
  wcast<<<192, 256, 0, stream>>>(dw, wf);
  dcn_sample<<<512, 256, 0, stream>>>(y, ow, ob, cols);
  dcn_gemm<<<8192, 256, 0, stream>>>(cols, wf, out);
}
